// Flashback_31499290149122
// MI455X (gfx1250) — compile-verified, matrix operand order checked by reading, not run on hardware
//
#include <hip/hip_runtime.h>
#include <hip/hip_bf16.h>
#include <math.h>

// Problem constants (from reference)
#define V_SZ 20000
#define H_SZ 128
#define B_SZ 64
#define S_SZ 200
#define BS_SZ (B_SZ * S_SZ)   // 12800
#define H2_SZ (2 * H_SZ)      // 256

typedef __attribute__((ext_vector_type(16))) __bf16 v16bf;
typedef __attribute__((ext_vector_type(8)))  __bf16 v8bf;
typedef __attribute__((ext_vector_type(8)))  float  v8f;

__device__ __forceinline__ v8f wmma_bf16(v16bf a, v16bf b, v8f c) {
  // D = A(16x32 bf16) * B(32x16 bf16) + C(16x16 f32)
  return __builtin_amdgcn_wmma_f32_16x16x32_bf16(false, a, false, b, (short)0, c,
                                                 false, false);
}

// 16-bit operand fragment for one K=32 chunk (ISA 7.12.2 layout):
// lanes 0-15 hold K = k0+{0..7, 16..23}, lanes 16-31 hold K = k0+{8..15, 24..31}
// rowbase = this lane's row (A: row M, B: row N of the [out,in] weight).
__device__ __forceinline__ v16bf frag_from_f32_row(const float* rowbase, int k0) {
  int lane = threadIdx.x & 31;
  int kh = (lane >> 4) << 3;
  const float* p = rowbase + k0 + kh;
  v16bf f;
#pragma unroll
  for (int i = 0; i < 8; i++) f[i] = (__bf16)p[i];
#pragma unroll
  for (int i = 0; i < 8; i++) f[8 + i] = (__bf16)p[16 + i];
  return f;
}

__device__ __forceinline__ v16bf frag_from_bf16_row(const __bf16* rowbase, int k0) {
  int lane = threadIdx.x & 31;
  int kh = (lane >> 4) << 3;
  const __bf16* p = rowbase + k0 + kh;
  v8bf lo = *(const v8bf*)(p);
  v8bf hi = *(const v8bf*)(p + 16);
  v16bf f;
#pragma unroll
  for (int i = 0; i < 8; i++) { f[i] = lo[i]; f[8 + i] = hi[i]; }
  return f;
}

// ---------------------------------------------------------------- K1: f32->bf16
__global__ void fb_cvt_bf16(const float* __restrict__ src, __bf16* __restrict__ dst,
                            int n) {
  int i = blockIdx.x * 256 + threadIdx.x;
  if (i < n) dst[i] = (__bf16)src[i];
}

// ------------------------------- K2: xW = emb[poi] @ W_ih^T + (b_ih + b_hh)
// grid = BS/16 blocks, 256 thr (8 waves; wave w = column tile w of H=128)
__global__ __launch_bounds__(256)
void fb_embed_gemm(const int* __restrict__ poi, const float* __restrict__ embW,
                   const float* __restrict__ Wih, const float* __restrict__ bih,
                   const float* __restrict__ bhh, float* __restrict__ xW) {
  int blk  = blockIdx.x;
  int wave = threadIdx.x >> 5;
  int lane = threadIdx.x & 31;
  int row  = blk * 16 + (lane & 15);            // flat b*S+s
  const float* arow = embW + (size_t)poi[row] * H_SZ;
  int n = wave * 16 + (lane & 15);
  const float* brow = Wih + (size_t)n * H_SZ;
  v8f acc = {};
#pragma unroll
  for (int kc = 0; kc < 4; kc++) {
    v16bf af = frag_from_f32_row(arow, kc * 32);
    v16bf bf = frag_from_f32_row(brow, kc * 32);
    acc = wmma_bf16(af, bf, acc);
  }
  float bias = bih[n] + bhh[n];
#pragma unroll
  for (int r = 0; r < 8; r++) {
    int m = blk * 16 + r + ((lane >> 4) << 3);
    xW[(size_t)m * H_SZ + n] = acc[r] + bias;
  }
}

// ------------------------- K3: sequential tanh RNN, single workgroup (1 WGP)
// h double-buffered in LDS (bf16); W_hh B-fragments pinned in VGPRs.
__global__ __launch_bounds__(256)
void fb_rnn(const float* __restrict__ xW, const float* __restrict__ Whh,
            float* __restrict__ rnnOut) {
  __shared__ __bf16 hbuf[2][B_SZ][H_SZ];        // 32 KB
  int tid  = threadIdx.x;
  int wave = tid >> 5;
  int lane = tid & 31;
  for (int i = tid; i < B_SZ * H_SZ; i += 256) ((__bf16*)hbuf[0])[i] = (__bf16)0.0f;

  int n = wave * 16 + (lane & 15);
  v16bf bw[4];
#pragma unroll
  for (int kc = 0; kc < 4; kc++) bw[kc] = frag_from_f32_row(Whh + (size_t)n * H_SZ, kc * 32);
  __syncthreads();

  int cur = 0;
  for (int t = 0; t < S_SZ; t++) {
#pragma unroll
    for (int rt = 0; rt < 4; rt++) {
      int mA = rt * 16 + (lane & 15);
      v8f acc = {};
#pragma unroll
      for (int kc = 0; kc < 4; kc++) {
        v16bf af = frag_from_bf16_row(&hbuf[cur][mA][0], kc * 32);
        acc = wmma_bf16(af, bw[kc], acc);
      }
#pragma unroll
      for (int r = 0; r < 8; r++) {
        int m = rt * 16 + r + ((lane >> 4) << 3);   // batch index 0..63
        float v = tanhf(acc[r] + xW[((size_t)m * S_SZ + t) * H_SZ + n]);
        hbuf[cur ^ 1][m][n] = (__bf16)v;
        rnnOut[((size_t)m * S_SZ + t) * H_SZ + n] = v;
      }
    }
    __syncthreads();
    cur ^= 1;
  }
}

// ------------- K4: spatio-temporal decay attention + concat user emb (VALU)
// grid = B*S blocks (one (b,i) row), 128 threads = H
__global__ __launch_bounds__(128)
void fb_attn(const float* __restrict__ ts, const float* __restrict__ loc,
             const float* __restrict__ rnnOut, const float* __restrict__ userW,
             const int* __restrict__ activeUser, float* __restrict__ outPuF32,
             __bf16* __restrict__ outPuB16) {
  int b = blockIdx.x / S_SZ;
  int i = blockIdx.x % S_SZ;
  int tid = threadIdx.x;
  __shared__ float wbuf[128];
  __shared__ float red[128];

  const float kOmega = 6.283185307179586f / 86400.0f;
  float ti  = ts[b * S_SZ + i];
  float six = loc[((size_t)b * S_SZ + i) * 2 + 0];
  float siy = loc[((size_t)b * S_SZ + i) * 2 + 1];

  float acc = 0.0f, spart = 0.0f;
  for (int j0 = 0; j0 <= i; j0 += 128) {
    int j = j0 + tid;
    float w = 0.0f;
    if (j <= i) {
      float dt = fmaxf(ti - ts[b * S_SZ + j], 0.0f);
      float dx = six - loc[((size_t)b * S_SZ + j) * 2 + 0];
      float dy = siy - loc[((size_t)b * S_SZ + j) * 2 + 1];
      float d2 = dx * dx + dy * dy;
      float ds = (d2 > 0.0f) ? sqrtf(d2) : 0.0f;
      float a = (cosf(dt * kOmega) + 1.0f) * 0.5f * expf(-dt * 0.01f);
      w = a * expf(-100.0f * ds) + 1e-10f;
      spart += w;
    }
    wbuf[tid] = w;
    __syncthreads();
    int lim = min(128, i - j0 + 1);
    for (int jj = 0; jj < lim; jj++)
      acc += wbuf[jj] * rnnOut[((size_t)(b * S_SZ + j0 + jj)) * H_SZ + tid];
    __syncthreads();
  }
  red[tid] = spart;
  __syncthreads();
  for (int off = 64; off > 0; off >>= 1) {
    if (tid < off) red[tid] += red[tid + off];
    __syncthreads();
  }
  float sw = fmaxf(red[0], 1e-10f);
  float ow = acc / sw;
  float pu = userW[(size_t)activeUser[b] * H_SZ + tid];
  size_t base = ((size_t)(b * S_SZ + i)) * H2_SZ;
  outPuF32[base + tid] = ow;
  outPuF32[base + H_SZ + tid] = pu;
  outPuB16[base + tid] = (__bf16)ow;
  outPuB16[base + H_SZ + tid] = (__bf16)pu;
}

// ------------------- K5: logits = out_pu @ fc_W^T + fc_b  (dominant GEMM)
// grid = BS/32 blocks (32 rows each); 8 waves stride over the 1250 N-tiles.
// Two M-tiles per block: every B-fragment load feeds two WMMAs (2x L2 reuse).
__global__ __launch_bounds__(256)
void fb_fc(const __bf16* __restrict__ apu, const __bf16* __restrict__ fcw,
           const float* __restrict__ fcb, float* __restrict__ y) {
  int blk  = blockIdx.x;
  int wave = threadIdx.x >> 5;
  int lane = threadIdx.x & 31;
  int m0   = blk * 32;
  v16bf a0[8], a1[8];
#pragma unroll
  for (int kc = 0; kc < 8; kc++) {
    a0[kc] = frag_from_bf16_row(apu + (size_t)(m0 + (lane & 15)) * H2_SZ, kc * 32);
    a1[kc] = frag_from_bf16_row(apu + (size_t)(m0 + 16 + (lane & 15)) * H2_SZ, kc * 32);
  }

  for (int nt = wave; nt < V_SZ / 16; nt += 8) {
    int n = nt * 16 + (lane & 15);
    const __bf16* brow = fcw + (size_t)n * H2_SZ;
    v8f acc0 = {}, acc1 = {};
#pragma unroll
    for (int kc = 0; kc < 8; kc++) {
      v16bf bf = frag_from_bf16_row(brow, kc * 32);
      acc0 = wmma_bf16(a0[kc], bf, acc0);
      acc1 = wmma_bf16(a1[kc], bf, acc1);
    }
    float bias = fcb[n];
#pragma unroll
    for (int r = 0; r < 8; r++) {
      int mr = m0 + r + ((lane >> 4) << 3);
      y[(size_t)mr * V_SZ + n]        = acc0[r] + bias;
      y[(size_t)(mr + 16) * V_SZ + n] = acc1[r] + bias;
    }
  }
}

// --------- K6: per-row online (max, sum of exp) over V=20000, float4 loads.
// Stores the RECIPROCAL of the sum so the normalize pass avoids v_div.
__global__ __launch_bounds__(256)
void fb_rowstat(const float* __restrict__ y, float* __restrict__ rmax,
                float* __restrict__ rinv) {
  int row = blockIdx.x;
  int tid = threadIdx.x;
  const float4* p = (const float4*)(y + (size_t)row * V_SZ);
  float m = -__builtin_inff(), s = 0.0f;
  for (int c = tid; c < V_SZ / 4; c += 256) {
    float4 v = p[c];
    float xs[4] = {v.x, v.y, v.z, v.w};
#pragma unroll
    for (int q = 0; q < 4; q++) {
      float nm = fmaxf(m, xs[q]);
      s = s * expf(m - nm) + expf(xs[q] - nm);
      m = nm;
    }
  }
  __shared__ float sm[256], ss[256];
  sm[tid] = m; ss[tid] = s;
  __syncthreads();
  for (int off = 128; off > 0; off >>= 1) {
    if (tid < off) {
      float m2 = sm[tid + off], s2 = ss[tid + off];
      float nm = fmaxf(sm[tid], m2);
      ss[tid] = ss[tid] * expf(sm[tid] - nm) + s2 * expf(m2 - nm);
      sm[tid] = nm;
    }
    __syncthreads();
  }
  if (tid == 0) { rmax[row] = sm[0]; rinv[row] = 1.0f / ss[0]; }
}

// --------------------- K7: y = exp(y - max) * inv_sum, float4 in/out
__global__ __launch_bounds__(256)
void fb_softmax(float* __restrict__ y, const float* __restrict__ rmax,
                const float* __restrict__ rinv) {
  int row = blockIdx.y;
  int c4 = blockIdx.x * 256 + threadIdx.x;
  if (c4 < V_SZ / 4) {
    float m = rmax[row], inv = rinv[row];
    float4* p = (float4*)(y + (size_t)row * V_SZ);
    float4 v = p[c4];
    v.x = expf(v.x - m) * inv;
    v.y = expf(v.y - m) * inv;
    v.z = expf(v.z - m) * inv;
    v.w = expf(v.w - m) * inv;
    p[c4] = v;
  }
}

extern "C" void kernel_launch(void* const* d_in, const int* in_sizes, int n_in,
                              void* d_out, int out_size, void* d_ws, size_t ws_size,
                              hipStream_t stream) {
  (void)in_sizes; (void)n_in; (void)out_size; (void)ws_size;
  const int*   poi      = (const int*)  d_in[0];
  /* d_in[1] lengths: all == S, unused */
  const float* tstamp   = (const float*)d_in[2];
  const float* loc      = (const float*)d_in[3];
  const int*   activeU  = (const int*)  d_in[4];
  const float* embW     = (const float*)d_in[5];
  const float* userW    = (const float*)d_in[6];
  const float* Wih      = (const float*)d_in[7];
  const float* Whh      = (const float*)d_in[8];
  const float* bih      = (const float*)d_in[9];
  const float* bhh      = (const float*)d_in[10];
  const float* fcW      = (const float*)d_in[11];
  const float* fcb      = (const float*)d_in[12];

  // Outputs: y [B,S,V] then out_pu [B,S,2H], both f32.
  float* y     = (float*)d_out;
  float* outPu = (float*)d_out + (size_t)BS_SZ * V_SZ;

  // Workspace layout (all offsets 256B aligned)
  char* ws = (char*)d_ws;
  const size_t szFcWb = (size_t)V_SZ * H2_SZ * sizeof(__bf16);   // 10,240,000
  const size_t szXW   = (size_t)BS_SZ * H_SZ * sizeof(float);    //  6,553,600
  const size_t szPuB  = (size_t)BS_SZ * H2_SZ * sizeof(__bf16);  //  6,553,600
  __bf16* fcWb   = (__bf16*)(ws);
  float*  xW     = (float*)(ws + szFcWb);
  float*  rnnOut = (float*)(ws + szFcWb + szXW);
  __bf16* puB16  = (__bf16*)(ws + szFcWb + 2 * szXW);
  float*  rmax   = (float*)(ws + szFcWb + 2 * szXW + szPuB);
  float*  rinv   = (float*)(ws + szFcWb + 2 * szXW + szPuB + BS_SZ * sizeof(float));

  int nFc = V_SZ * H2_SZ;
  fb_cvt_bf16<<<(nFc + 255) / 256, 256, 0, stream>>>(fcW, fcWb, nFc);
  fb_embed_gemm<<<BS_SZ / 16, 256, 0, stream>>>(poi, embW, Wih, bih, bhh, xW);
  fb_rnn<<<1, 256, 0, stream>>>(xW, Whh, rnnOut);
  fb_attn<<<BS_SZ, 128, 0, stream>>>(tstamp, loc, rnnOut, userW, activeU, outPu, puB16);
  fb_fc<<<BS_SZ / 32, 256, 0, stream>>>(puB16, fcWb, fcb, y);
  fb_rowstat<<<BS_SZ, 256, 0, stream>>>(y, rmax, rinv);
  fb_softmax<<<dim3((V_SZ / 4 + 255) / 256, BS_SZ), 256, 0, stream>>>(y, rmax, rinv);
}
